// NeuromorphicEnergyOptimizer_25288767439433
// MI455X (gfx1250) — matrix operand order repaired and checked by reading, not compiled, not run on hardware
//
#include <hip/hip_runtime.h>
#include <hip/hip_bf16.h>

typedef __attribute__((ext_vector_type(16))) _Float16 v16h;
typedef __attribute__((ext_vector_type(8)))  _Float16 v8h;
typedef __attribute__((ext_vector_type(4)))  _Float16 v4h;
typedef __attribute__((ext_vector_type(8)))  float    v8f;

// ---------------------------------------------------------------------------
// f32 -> f16 conversion (vectorized x4)
// ---------------------------------------------------------------------------
__global__ void cvt_f32_to_f16(const float* __restrict__ in,
                               _Float16* __restrict__ out, int n) {
  int i = (blockIdx.x * blockDim.x + threadIdx.x) * 4;
  if (i < n) {
    float4 v = *(const float4*)(in + i);
    v4h o = { (_Float16)v.x, (_Float16)v.y, (_Float16)v.z, (_Float16)v.w };
    *(v4h*)(out + i) = o;
  }
}

// ---------------------------------------------------------------------------
// Leaky-integrate-and-fire scan: each thread owns one (batch, neuron) lane.
//   reset = H(mem_old - 1);  mem = beta*mem_old + cur - reset;  spk = H(mem-1)
// ---------------------------------------------------------------------------
__global__ void leaky_scan(const float* __restrict__ cur,
                           _Float16* __restrict__ spk,
                           float beta, int T, int BH) {
  int idx = blockIdx.x * blockDim.x + threadIdx.x;
  if (idx >= BH) return;
  float mem = 0.0f;
  for (int t = 0; t < T; ++t) {
    float c = cur[(size_t)t * BH + idx];
    float reset = (mem > 1.0f) ? 1.0f : 0.0f;
    mem = beta * mem + c - reset;
    spk[(size_t)t * BH + idx] = (_Float16)((mem > 1.0f) ? 1.0f : 0.0f);
  }
}

// ---------------------------------------------------------------------------
// A-fragment loader (ISA 7.12.2, 16-bit A 16x32, wave32):
//   lane<16 : row m0+lane,    halves {k0..k0+7, k0+16..k0+23}
//   lane>=16: row m0+lane-16, halves {k0+8..k0+15, k0+24..k0+31}
// ---------------------------------------------------------------------------
__device__ __forceinline__ v16h load_a_frag(const _Float16* __restrict__ A,
                                            int lda, int m0, int k0,
                                            int lm, bool hi) {
  const _Float16* p = A + (size_t)(m0 + lm) * lda + k0 + (hi ? 8 : 0);
  v8h lo = *(const v8h*)p;
  v8h hh = *(const v8h*)(p + 16);
  v16h r;
#pragma unroll
  for (int i = 0; i < 8; ++i) { r[i] = lo[i]; r[i + 8] = hh[i]; }
  return r;
}

// ---------------------------------------------------------------------------
// GEMM: C[M,N] = A[M,K] @ W[N,K]^T + bias  (f16 in, f32 accumulate/out)
// Block = 4 waves (128 thr). Block tile 128x128:
//   wave w -> rows [bx*128 + w*32, +32) as 2 M-subtiles,
//   8 N-subtiles, 16 accumulators per wave.
// B chunk (128 rows x 32 halves) staged in LDS, double-buffered, shared by
// all 4 waves (B fragment per lane: W[n0+lm][k0 + (hi?8:0) .. +7, +16 .. +23]).
// LDS rows padded to 40 halves -> conflict-free stride-20-dword reads.
// ---------------------------------------------------------------------------
#define GEMM_MT 2
#define GEMM_NT 8
#define NBLK    128
#define LDS_STRIDE 40

__global__ __launch_bounds__(128) void gemm_f16_wmma(
    const _Float16* __restrict__ A, const _Float16* __restrict__ W,
    const float* __restrict__ bias, float* __restrict__ C,
    int M, int N, int K) {
  __shared__ _Float16 bsh[2][NBLK * LDS_STRIDE];

  const int tid  = threadIdx.x;
  const int wave = tid >> 5;
  const int lane = tid & 31;
  const int lm   = lane & 15;
  const bool hi  = lane >= 16;
  const int m0 = (blockIdx.x * 4 + wave) * 32;
  const int n0 = blockIdx.y * NBLK;

  v8f acc[GEMM_MT][GEMM_NT] = {};

  // Each thread owns one B row of the block's N-range for staging.
  const _Float16* wrow = W + (size_t)(n0 + tid) * K;

  { // stage chunk 0 into buffer 0
    v8h s0 = *(const v8h*)(wrow);
    v8h s1 = *(const v8h*)(wrow + 8);
    v8h s2 = *(const v8h*)(wrow + 16);
    v8h s3 = *(const v8h*)(wrow + 24);
    _Float16* d = &bsh[0][tid * LDS_STRIDE];
    *(v8h*)(d) = s0; *(v8h*)(d + 8) = s1;
    *(v8h*)(d + 16) = s2; *(v8h*)(d + 24) = s3;
  }
  __syncthreads();

  for (int k0 = 0; k0 < K; k0 += 32) {
    const int cur  = (k0 >> 5) & 1;
    const bool more = (k0 + 32) < K;

    // kick off next chunk's global B loads early (hide behind WMMAs)
    v8h s0, s1, s2, s3;
    if (more) {
      const _Float16* p = wrow + k0 + 32;
      s0 = *(const v8h*)(p);      s1 = *(const v8h*)(p + 8);
      s2 = *(const v8h*)(p + 16); s3 = *(const v8h*)(p + 24);
      __builtin_prefetch(A + (size_t)(m0 + lm) * K + k0 + 32, 0, 3);
    }

    // A fragments straight from global (no intra-block reuse exists)
    v16h af[GEMM_MT];
#pragma unroll
    for (int mi = 0; mi < GEMM_MT; ++mi)
      af[mi] = load_a_frag(A, K, m0 + mi * 16, k0, lm, hi);

    // compute: 8 N-subtiles x 2 M-subtiles = 16 WMMAs per chunk
#pragma unroll
    for (int j = 0; j < GEMM_NT; ++j) {
      const _Float16* bp =
          &bsh[cur][(j * 16 + lm) * LDS_STRIDE + (hi ? 8 : 0)];
      v8h blo = *(const v8h*)bp;
      v8h bhh = *(const v8h*)(bp + 16);
      v16h bf;
#pragma unroll
      for (int i = 0; i < 8; ++i) { bf[i] = blo[i]; bf[i + 8] = bhh[i]; }
#pragma unroll
      for (int mi = 0; mi < GEMM_MT; ++mi)
        acc[mi][j] = __builtin_amdgcn_wmma_f32_16x16x32_f16(
            /*neg_a=*/false, af[mi], /*neg_b=*/false, bf,
            /*c_mod=*/(short)0, acc[mi][j],
            /*reuse_a=*/false, /*reuse_b=*/false);
    }

    // land next chunk into the alternate buffer
    if (more) {
      _Float16* d = &bsh[cur ^ 1][tid * LDS_STRIDE];
      *(v8h*)(d) = s0; *(v8h*)(d + 8) = s1;
      *(v8h*)(d + 16) = s2; *(v8h*)(d + 24) = s3;
    }
    __syncthreads();
  }

  // C/D layout: VGPR r -> (M = m0 + mi*16 + r + (hi?8:0), N = n0 + j*16 + lm)
#pragma unroll
  for (int mi = 0; mi < GEMM_MT; ++mi) {
#pragma unroll
    for (int j = 0; j < GEMM_NT; ++j) {
      int n = n0 + j * 16 + lm;
      float bv = bias[n];
#pragma unroll
      for (int r = 0; r < 8; ++r) {
        int m = m0 + mi * 16 + r + (hi ? 8 : 0);
        C[(size_t)m * N + n] = acc[mi][j][r] + bv;
      }
    }
  }
}

// ---------------------------------------------------------------------------
// Launch: 3 GEMMs + 2 scans + conversions, all stream-ordered.
// ---------------------------------------------------------------------------
extern "C" void kernel_launch(void* const* d_in, const int* in_sizes, int n_in,
                              void* d_out, int out_size, void* d_ws, size_t ws_size,
                              hipStream_t stream) {
  const float* x  = (const float*)d_in[0];
  const float* w1 = (const float*)d_in[1];
  const float* b1 = (const float*)d_in[2];
  const float* w2 = (const float*)d_in[3];
  const float* b2 = (const float*)d_in[4];
  const float* w3 = (const float*)d_in[5];
  const float* b3 = (const float*)d_in[6];

  constexpr int T = 256, Bb = 64, I = 512, H1 = 1024, H2 = 1024, O = 256;
  constexpr int M = T * Bb;  // 16384 rows in every GEMM

  // Workspace layout (stream-ordered reuse):
  //   [0,   64MB)  curbuf f32[M,1024]   : cur1 then cur2
  //   [64,  96MB)  spk/xh region        : xh f16[M,I] (dead after GEMM1),
  //                                       then spk1 f16[M,H1], then spk2
  //   [96, ~99.5MB) w1h, w2h, w3h f16
  char* ws = (char*)d_ws;
  float*    curbuf = (float*)ws;
  char*     shr    = ws + (size_t)M * H1 * sizeof(float);
  _Float16* spkbuf = (_Float16*)shr;
  _Float16* xh     = (_Float16*)shr;  // aliases spkbuf (dead before scan1 writes)
  char*     wreg   = shr + (size_t)M * H1 * sizeof(_Float16);
  _Float16* w1h = (_Float16*)wreg;
  _Float16* w2h = w1h + (size_t)H1 * I;
  _Float16* w3h = w2h + (size_t)H2 * H1;

  const int CT = 256;
  cvt_f32_to_f16<<<dim3((M * I / 4 + CT - 1) / CT), dim3(CT), 0, stream>>>(x, xh, M * I);
  cvt_f32_to_f16<<<dim3((H1 * I / 4 + CT - 1) / CT), dim3(CT), 0, stream>>>(w1, w1h, H1 * I);
  cvt_f32_to_f16<<<dim3((H2 * H1 / 4 + CT - 1) / CT), dim3(CT), 0, stream>>>(w2, w2h, H2 * H1);
  cvt_f32_to_f16<<<dim3((O * H2 / 4 + CT - 1) / CT), dim3(CT), 0, stream>>>(w3, w3h, O * H2);

  dim3 gblk(128);
  // Layer 1 GEMM over all timesteps at once
  gemm_f16_wmma<<<dim3(M / 128, H1 / NBLK), gblk, 0, stream>>>(
      xh, w1h, b1, curbuf, M, H1, I);
  // Layer-1 membrane scan (65536 independent lanes over T)
  leaky_scan<<<dim3(Bb * H1 / CT), dim3(CT), 0, stream>>>(curbuf, spkbuf, 0.9f, T, Bb * H1);
  // Layer 2 GEMM (spikes exact in f16)
  gemm_f16_wmma<<<dim3(M / 128, H2 / NBLK), gblk, 0, stream>>>(
      spkbuf, w2h, b2, curbuf, M, H2, H1);
  // Layer-2 membrane scan
  leaky_scan<<<dim3(Bb * H2 / CT), dim3(CT), 0, stream>>>(curbuf, spkbuf, 0.85f, T, Bb * H2);
  // Output GEMM straight into d_out [T,B,O]
  gemm_f16_wmma<<<dim3(M / 128, O / NBLK), gblk, 0, stream>>>(
      spkbuf, w3h, b3, (float*)d_out, M, O, H2);
}
